// Wavenet_17600775979793
// MI455X (gfx1250) — matrix operand order repaired and checked by reading, not compiled, not run on hardware
//
#include <hip/hip_runtime.h>

// ---------------------------------------------------------------------------
// WaveNet inference for MI455X (gfx1250, wave32, WMMA f16 -> f32 accum).
// v4: B operand staged in LDS (conflict-free padded layout), gate computed
//     in registers via row-pair wave assignment (no xc LDS round trip),
//     weight A-fragments hoisted across n-tiles (16 loads -> 64 WMMAs).
// ---------------------------------------------------------------------------

typedef __attribute__((ext_vector_type(16))) _Float16 v16h;
typedef __attribute__((ext_vector_type(8)))  _Float16 v8h;
typedef __attribute__((ext_vector_type(8)))  float    v8f;

#define T_LEN  16384
#define B_N    4
#define IN_C   256
#define RES_C  128
#define SKP_C  256
#define CDT_C  80
#define OUT_C  256
#define NLAY   20
#define DRATE  64
#define NTILE  64     // time columns per workgroup (== DRATE so cond is constant)
#define LDR    136    // padded LDS row stride for 128-wide rows (bank spread)
#define LDP    264    // padded LDS row stride for 256-wide rows

// ---- WMMA fragment helpers (CDNA5 wave32 layouts, 16x16x32 f16) ------------
__device__ inline v16h load_A16(const _Float16* __restrict__ W, int ldk, int lane) {
    const _Float16* p = W + (size_t)(lane & 15) * ldk + ((lane >> 4) << 4);
    v16h a;
    ((v8h*)&a)[0] = *(const v8h*)p;
    ((v8h*)&a)[1] = *(const v8h*)(p + 8);
    return a;
}
__device__ inline v16h load_B16(const _Float16* __restrict__ act, int ldc, int lane) {
    return *(const v16h*)(act + (size_t)(lane & 15) * ldc + ((lane >> 4) << 4));
}
// B fragment from LDS: 16 contiguous halfs as two 16B-aligned v8h reads.
__device__ inline v16h load_B_lds(const _Float16* p) {
    v16h r;
    ((v8h*)&r)[0] = *(const v8h*)p;
    ((v8h*)&r)[1] = *(const v8h*)(p + 8);
    return r;
}
__device__ inline v8f wmma_f16(v16h a, v16h b, v8f c) {
    return __builtin_amdgcn_wmma_f32_16x16x32_f16(false, a, false, b, (short)0, c,
                                                  false, false);
}
__device__ inline float fast_tanh(float x) {
    float e = __expf(2.0f * x);
    return (e - 1.0f) * __builtin_amdgcn_rcpf(e + 1.0f);
}
__device__ inline float fast_sigmoid(float x) {
    return __builtin_amdgcn_rcpf(1.0f + __expf(-x));
}

// ---- Prep: convert/rearrange weights to f16 row-major [M][K] ---------------
//   Wfirst [2][128][256]  @0        Wdil [20][2][256][128] @65536
//   Wres   [20][128][128] @1376256  Wskp [20][256][128]    @1703936
//   Wlc1   [256][256]     @2359296  Wlc2 [256][256]        @2424832
__global__ void k_prep_w(const float* __restrict__ fw, const float* __restrict__ dw,
                         const float* __restrict__ rw, const float* __restrict__ sw,
                         const float* __restrict__ l1, const float* __restrict__ l2,
                         _Float16* __restrict__ W) {
    int idx = blockIdx.x * blockDim.x + threadIdx.x;
    if (idx < 65536) {
        int tap = idx >> 15, o = (idx >> 8) & 127, k = idx & 255;
        W[idx] = (_Float16)fw[(o * 256 + k) * 2 + tap];
        return;
    }
    idx -= 65536;
    if (idx < 1310720) {
        int k = idx & 127, o = (idx >> 7) & 255, tap = (idx >> 15) & 1, i = idx >> 16;
        W[65536 + idx] = (_Float16)dw[(((size_t)i * 256 + o) * 128 + k) * 2 + tap];
        return;
    }
    idx -= 1310720;
    if (idx < 327680) { W[1376256 + idx] = (_Float16)rw[idx]; return; }
    idx -= 327680;
    if (idx < 655360) { W[1703936 + idx] = (_Float16)sw[idx]; return; }
    idx -= 655360;
    if (idx < 65536)  { W[2359296 + idx] = (_Float16)l1[idx]; return; }
    idx -= 65536;
    if (idx < 65536)  { W[2424832 + idx] = (_Float16)l2[idx]; return; }
}

// ---- Prep: x (B,IN,T) f32 -> xT [b][t][c] f16 ------------------------------
__global__ void k_prep_x(const float* __restrict__ x, _Float16* __restrict__ xT) {
    int idx = blockIdx.x * blockDim.x + threadIdx.x;      // 2^24 total
    int c = idx & 255;
    int t = (idx >> 8) & 16383;
    int b = idx >> 22;
    xT[idx] = (_Float16)x[((size_t)b * 256 + c) * 16384 + t];
}

// ---- Conditioning: cond[i][b][tc][o] = cdt_w[i] @ c[b,:,tc] + cdt_b[i] -----
__global__ void k_cond(const float* __restrict__ c, const float* __restrict__ cw,
                       const float* __restrict__ cb, float* __restrict__ cond) {
    int idx = blockIdx.x * blockDim.x + threadIdx.x;
    if (idx >= NLAY * B_N * (T_LEN / DRATE) * (2 * RES_C)) return;
    int o  = idx & 255;
    int tc = (idx >> 8) & 255;
    int b  = (idx >> 16) & 3;
    int i  = idx >> 18;
    const float* wp = cw + ((size_t)i * 2 * RES_C + o) * CDT_C;
    const float* cp = c + (size_t)b * CDT_C * (T_LEN / DRATE) + tc;
    float s = cb[i * 2 * RES_C + o];
    for (int k = 0; k < CDT_C; k++) s += wp[k] * cp[(size_t)k * (T_LEN / DRATE)];
    cond[idx] = s;
}

// ---- First layer: res0 = tanh(W0@x[t-1] + W1@x[t] + b) ---------------------
__global__ __launch_bounds__(256)
void k_first(const _Float16* __restrict__ xT, const _Float16* __restrict__ Wf,
             const float* __restrict__ fb, _Float16* __restrict__ resOut,
             const _Float16* __restrict__ zpage) {
    int b = blockIdx.y;
    int t0 = blockIdx.x * NTILE;
    int wave = threadIdx.x >> 5, lane = threadIdx.x & 31;
    const _Float16* xb = xT + (size_t)b * T_LEN * IN_C;
    _Float16* ro = resOut + (size_t)b * T_LEN * RES_C;

    for (int tile = wave; tile < 32; tile += 8) {         // M=128 (8) x N=64 (4)
        int m0 = (tile >> 2) * 16;
        int n0 = t0 + (tile & 3) * 16;
        const _Float16* W1 = Wf + (size_t)RES_C * IN_C + (size_t)m0 * IN_C;
        const _Float16* W0 = Wf + (size_t)m0 * IN_C;
        int tcol = n0 + (lane & 15) - 1;
        const _Float16* base = xb + (size_t)tcol * IN_C + ((lane >> 4) << 4);
        const _Float16* bsrc = (tcol >= 0) ? base : zpage;

        v8f acc = {};
        #pragma unroll
        for (int s = 0; s < 16; s++) {
            const _Float16* Wp = (s < 8) ? (W1 + s * 32) : (W0 + (s - 8) * 32);
            v16h a = load_A16(Wp, IN_C, lane);
            v16h bv = (s < 8)
                ? load_B16(xb + (size_t)n0 * IN_C + s * 32, IN_C, lane)
                : *(const v16h*)(bsrc + (s - 8) * 32);
            acc = wmma_f16(a, bv, acc);
        }
        int t = n0 + (lane & 15);
        int r0 = m0 + ((lane >> 4) << 3);
        v8h ov;
        for (int v = 0; v < 8; v++) ov[v] = (_Float16)fast_tanh(acc[v] + fb[r0 + v]);
        *(v8h*)(ro + (size_t)t * RES_C + r0) = ov;
    }
}

// ---- Dilated layer: LDS-staged conv + in-register gate + skip/res GEMMs ----
__global__ __launch_bounds__(256)
void k_layer(const _Float16* __restrict__ resIn, _Float16* __restrict__ resOut,
             float* __restrict__ skip, const _Float16* __restrict__ Wd,
             const _Float16* __restrict__ Wr, const _Float16* __restrict__ Ws,
             const float* __restrict__ db, const float* __restrict__ rb,
             const float* __restrict__ sb, const float* __restrict__ cond,
             const _Float16* __restrict__ zpage, int dil, int last) {
    __shared__ _Float16 cur_s [NTILE * LDR];   // [n][c] current columns   17KB
    __shared__ _Float16 past_s[NTILE * LDR];   // [n][c] columns t-dil     17KB
    __shared__ _Float16 gate_s[NTILE * LDR];   // [n][r] gate              17KB
    int b = blockIdx.y;
    int t0 = blockIdx.x * NTILE;
    int wave = threadIdx.x >> 5, lane = threadIdx.x & 31;
    const _Float16* ri = resIn + (size_t)b * T_LEN * RES_C;
    const float* cnd = cond + ((size_t)b * (T_LEN / DRATE) + blockIdx.x) * (2 * RES_C);

    // Stage both residual windows (coalesced 16B chunks, zero-page causal pad)
    for (int idx = threadIdx.x; idx < 1024; idx += 256) {  // 64 cols x 16 chunks
        int n = idx >> 4;
        int co = (idx & 15) << 3;
        *(v8h*)&cur_s[n * LDR + co] =
            *(const v8h*)(ri + (size_t)(t0 + n) * RES_C + co);
        int tp = t0 + n - dil;
        const _Float16* s0 = (tp >= 0) ? (ri + (size_t)tp * RES_C + co)
                                       : (zpage + co);
        *(v8h*)&past_s[n * LDR + co] = *(const v8h*)s0;
    }
    __syncthreads();

    // Phase 1: wave owns row pair (m0a, m0a+128) for all 4 n-tiles.
    // accA holds xa rows, accB holds xb rows with identical (lane,v) mapping,
    // so the tanh*sigmoid gate is computed entirely in registers.
    int m0a = wave * 16;
    v16h af[16];
    #pragma unroll
    for (int s = 0; s < 4; s++) {
        af[s]      = load_A16(Wd + 32768 + (size_t)m0a * RES_C + s * 32, RES_C, lane);
        af[s + 4]  = load_A16(Wd +         (size_t)m0a * RES_C + s * 32, RES_C, lane);
        af[s + 8]  = load_A16(Wd + 32768 + (size_t)(m0a + 128) * RES_C + s * 32,
                              RES_C, lane);
        af[s + 12] = load_A16(Wd +         (size_t)(m0a + 128) * RES_C + s * 32,
                              RES_C, lane);
    }
    int r0 = m0a + ((lane >> 4) << 3);
    for (int nt = 0; nt < 4; nt++) {
        int n0l = nt * 16;
        const _Float16* bc = &cur_s [(n0l + (lane & 15)) * LDR + ((lane >> 4) << 4)];
        const _Float16* bp = &past_s[(n0l + (lane & 15)) * LDR + ((lane >> 4) << 4)];
        v8f accA = {}, accB = {};
        #pragma unroll
        for (int s = 0; s < 4; s++) {
            v16h bv = load_B_lds(bc + s * 32);
            accA = wmma_f16(af[s],     bv, accA);
            accB = wmma_f16(af[s + 8], bv, accB);
        }
        #pragma unroll
        for (int s = 0; s < 4; s++) {
            v16h bv = load_B_lds(bp + s * 32);
            accA = wmma_f16(af[s + 4],  bv, accA);
            accB = wmma_f16(af[s + 12], bv, accB);
        }
        int n = n0l + (lane & 15);
        v8h gv;
        for (int v = 0; v < 8; v++) {
            float xa = accA[v] + db[r0 + v]       + cnd[r0 + v];
            float xb = accB[v] + db[r0 + v + 128] + cnd[r0 + v + 128];
            gv[v] = (_Float16)(fast_tanh(xa) * fast_sigmoid(xb));
        }
        *(v8h*)&gate_s[n * LDR + r0] = gv;
    }
    __syncthreads();

    // Phase 2: skip += Ws@gate + sb (jm=0,1) ; res = resIn + Wr@gate + rb (jm=2)
    for (int jm = 0; jm < 3; jm++) {
        if (jm == 2 && last) break;
        bool isSkip = jm < 2;
        int m0 = isSkip ? (wave + jm * 8) * 16 : wave * 16;
        const _Float16* Wp = (isSkip ? Ws : Wr) + (size_t)m0 * RES_C;
        v16h aw[4];
        #pragma unroll
        for (int s = 0; s < 4; s++) aw[s] = load_A16(Wp + s * 32, RES_C, lane);
        int rr0 = m0 + ((lane >> 4) << 3);
        for (int nt = 0; nt < 4; nt++) {
            int n0l = nt * 16;
            const _Float16* gb = &gate_s[(n0l + (lane & 15)) * LDR +
                                         ((lane >> 4) << 4)];
            v8f acc = {};
            #pragma unroll
            for (int s = 0; s < 4; s++)
                acc = wmma_f16(aw[s], load_B_lds(gb + s * 32), acc);
            int t = t0 + n0l + (lane & 15);
            if (isSkip) {
                float* sp = skip + ((size_t)b * T_LEN + t) * SKP_C + rr0;
                for (int v = 0; v < 8; v++) sp[v] += acc[v] + sb[rr0 + v];
            } else {
                const _Float16* rin = ri + (size_t)t * RES_C + rr0;
                v8h old = *(const v8h*)rin, ov;
                for (int v = 0; v < 8; v++)
                    ov[v] = (_Float16)((float)old[v] + acc[v] + rb[rr0 + v]);
                *(v8h*)(resOut + ((size_t)b * T_LEN + t) * RES_C + rr0) = ov;
            }
        }
    }
}

// ---- Post: out = lc2 @ relu(lc1 @ relu(skip)) ------------------------------
__global__ __launch_bounds__(256)
void k_post(const float* __restrict__ skip, const _Float16* __restrict__ W1,
            const _Float16* __restrict__ W2, float* __restrict__ out) {
    __shared__ _Float16 h_s[NTILE * LDP];   // [n][s] padded, ~33KB (reused for h2)
    int b = blockIdx.y;
    int t0 = blockIdx.x * NTILE;
    int wave = threadIdx.x >> 5, lane = threadIdx.x & 31;

    const float* sp = skip + ((size_t)b * T_LEN + t0) * SKP_C;
    for (int idx = threadIdx.x; idx < NTILE * SKP_C; idx += 256) {
        int n = idx >> 8, s = idx & 255;
        float v = sp[idx];
        h_s[n * LDP + s] = (_Float16)(v > 0.f ? v : 0.f);
    }
    __syncthreads();

    v8f acc[8];
    for (int jm = 0; jm < 2; jm++) {          // h2 = W1 @ h, wave owns 2 m-tiles
        int m0 = (wave + jm * 8) * 16;
        v16h aw[8];
        #pragma unroll
        for (int s = 0; s < 8; s++)
            aw[s] = load_A16(W1 + (size_t)m0 * SKP_C + s * 32, SKP_C, lane);
        for (int nt = 0; nt < 4; nt++) {
            const _Float16* hb = &h_s[(nt * 16 + (lane & 15)) * LDP +
                                      ((lane >> 4) << 4)];
            v8f a = {};
            #pragma unroll
            for (int s = 0; s < 8; s++)
                a = wmma_f16(aw[s], load_B_lds(hb + s * 32), a);
            acc[jm * 4 + nt] = a;
        }
    }
    __syncthreads();
    for (int jm = 0; jm < 2; jm++) {          // relu(h2) back into LDS
        int m0 = (wave + jm * 8) * 16;
        int r0 = m0 + ((lane >> 4) << 3);
        for (int nt = 0; nt < 4; nt++) {
            int n = nt * 16 + (lane & 15);
            for (int v = 0; v < 8; v++) {
                float y = acc[jm * 4 + nt][v];
                h_s[n * LDP + r0 + v] = (_Float16)(y > 0.f ? y : 0.f);
            }
        }
    }
    __syncthreads();
    for (int jm = 0; jm < 2; jm++) {          // out = W2 @ h2 -> (B,OUT,T)
        int m0 = (wave + jm * 8) * 16;
        v16h aw[8];
        #pragma unroll
        for (int s = 0; s < 8; s++)
            aw[s] = load_A16(W2 + (size_t)m0 * SKP_C + s * 32, SKP_C, lane);
        int r0 = m0 + ((lane >> 4) << 3);
        for (int nt = 0; nt < 4; nt++) {
            const _Float16* hb = &h_s[(nt * 16 + (lane & 15)) * LDP +
                                      ((lane >> 4) << 4)];
            v8f a = {};
            #pragma unroll
            for (int s = 0; s < 8; s++)
                a = wmma_f16(aw[s], load_B_lds(hb + s * 32), a);
            int t = t0 + nt * 16 + (lane & 15);
            for (int v = 0; v < 8; v++)
                out[((size_t)b * OUT_C + r0 + v) * T_LEN + t] = a[v];
        }
    }
}

// ---------------------------------------------------------------------------
extern "C" void kernel_launch(void* const* d_in, const int* in_sizes, int n_in,
                              void* d_out, int out_size, void* d_ws, size_t ws_size,
                              hipStream_t stream) {
    const float* x         = (const float*)d_in[0];
    const float* c         = (const float*)d_in[1];
    const float* first_w   = (const float*)d_in[2];
    const float* first_b   = (const float*)d_in[3];
    const float* dilated_w = (const float*)d_in[4];
    const float* dilated_b = (const float*)d_in[5];
    const float* cdt_w     = (const float*)d_in[6];
    const float* cdt_b     = (const float*)d_in[7];
    const float* res_w     = (const float*)d_in[8];
    const float* res_b     = (const float*)d_in[9];
    const float* skp_w     = (const float*)d_in[10];
    const float* skp_b     = (const float*)d_in[11];
    const float* lc1_w     = (const float*)d_in[12];
    const float* lc2_w     = (const float*)d_in[13];
    float* out = (float*)d_out;

    char* ws = (char*)d_ws;
    size_t o_xT   = 0;                         // f16 [B][T][IN]   33554432
    size_t o_resA = o_xT   + 33554432;         // f16 [B][T][RES]  16777216
    size_t o_resB = o_resA + 16777216;
    size_t o_skip = o_resB + 16777216;         // f32 [B][T][SKP]  67108864
    size_t o_cond = o_skip + 67108864;         // f32 [NL][B][256][256]
    size_t o_w    = o_cond + 20971520;
    size_t o_z    = o_w    + 4980736;          // 1KB zero page

    _Float16* xT   = (_Float16*)(ws + o_xT);
    _Float16* resA = (_Float16*)(ws + o_resA);
    _Float16* resB = (_Float16*)(ws + o_resB);
    float*    skip = (float*)(ws + o_skip);
    float*    cond = (float*)(ws + o_cond);
    _Float16* W    = (_Float16*)(ws + o_w);
    _Float16* zpage= (_Float16*)(ws + o_z);

    _Float16* Wfirst = W;
    _Float16* Wdil   = W + 65536;
    _Float16* Wres   = W + 1376256;
    _Float16* Wskp   = W + 1703936;
    _Float16* Wlc1   = W + 2359296;
    _Float16* Wlc2   = W + 2424832;

    hipMemsetAsync(skip, 0, (size_t)B_N * T_LEN * SKP_C * sizeof(float), stream);
    hipMemsetAsync(zpage, 0, 1024, stream);
    k_prep_w<<<2490368 / 256, 256, 0, stream>>>(first_w, dilated_w, res_w, skp_w,
                                                lc1_w, lc2_w, W);
    k_prep_x<<<16777216 / 256, 256, 0, stream>>>(x, xT);
    k_cond<<<5242880 / 256, 256, 0, stream>>>(c, cdt_w, cdt_b, cond);

    dim3 grid(T_LEN / NTILE, B_N);   // 256 x 4 workgroups, 256 threads (8 waves)
    k_first<<<grid, 256, 0, stream>>>(xT, Wfirst, first_b, resA, zpage);

    _Float16* rin = resA;
    _Float16* rout = resB;
    for (int i = 0; i < NLAY; i++) {
        int dil = 1 << (i % (NLAY / 2));
        int last = (i == NLAY - 1) ? 1 : 0;
        k_layer<<<grid, 256, 0, stream>>>(
            rin, rout, skip,
            Wdil + (size_t)i * 65536, Wres + (size_t)i * 16384,
            Wskp + (size_t)i * 32768,
            dilated_b + (size_t)i * 2 * RES_C, res_b + (size_t)i * RES_C,
            skp_b + (size_t)i * SKP_C,
            cond + (size_t)i * B_N * (T_LEN / DRATE) * 2 * RES_C,
            zpage, dil, last);
        _Float16* tmp = rin; rin = rout; rout = tmp;
    }

    k_post<<<grid, 256, 0, stream>>>(skip, Wlc1, Wlc2, out);
}
